// TemporalAttentionBlock_60516089200972
// MI455X (gfx1250) — matrix operand order repaired and checked by reading, not compiled, not run on hardware
//
#include <hip/hip_runtime.h>
#include <hip/hip_bf16.h>

// ---------------------------------------------------------------------------
// Types for CDNA5 WMMA (wave32): V_WMMA_F32_16X16X32_BF16
// ---------------------------------------------------------------------------
typedef __attribute__((ext_vector_type(16))) __bf16 bf16x16;
typedef __attribute__((ext_vector_type(8)))  float  f32x8;
typedef __attribute__((ext_vector_type(4)))  unsigned int u32x4;
typedef __attribute__((ext_vector_type(8)))  int          i32x8;
typedef __attribute__((ext_vector_type(4)))  int          i32x4;

union FragB16 { bf16x16 v; uint4 u[2]; };

#define EPI_F32  0
#define EPI_GELU 1
#define EPI_QKV  2

// Problem constants
#define BSZ 128
#define SEQ 64
#define DIM 1024
#define NH  16
#define HD  64
#define FFN 4096
#define MTOK (BSZ*SEQ)               // 8192 rows
#define QSZ ((size_t)BSZ*NH*SEQ*HD)  // per q/k/v tensor elements

__device__ __forceinline__ float gelu_exact(float x) {
    return 0.5f * x * (1.0f + erff(x * 0.70710678118654752f));
}

// ---------------------------------------------------------------------------
// CDNA5 async global -> LDS copy (ASYNCcnt-tracked), GVS addressing.
// ---------------------------------------------------------------------------
__device__ __forceinline__ void async_copy_b128(uint32_t lds_off,
                                                const void* sbase,
                                                uint32_t voff_bytes) {
    asm volatile("global_load_async_to_lds_b128 %0, %1, %2"
                 :
                 : "v"(lds_off), "v"(voff_bytes), "s"(sbase)
                 : "memory");
}
__device__ __forceinline__ void wait_async0() {
    asm volatile("s_wait_asynccnt 0x0" ::: "memory");
}
// LDS byte offset of a __shared__ object (flat LDS aperture: low 32 bits)
__device__ __forceinline__ uint32_t lds_offset_of(const void* p) {
    return (uint32_t)(uintptr_t)p;
}

// ---------------------------------------------------------------------------
// CDNA5 Tensor Data Mover: 2D tile (tile_d0 x tile_d1 elements of 2 bytes)
// from global (row stride = stride_elems) into LDS at lds_addr.
// LDS padding: +4 DWORDs every 16 DWORDs -> row stride 40 bf16 (matches LDT).
// Descriptor bit layout per CDNA5 ISA ch.8 (D# groups 0/1; groups 2/3 zero).
// amdgpu-toolchain (clang-23) builtin arity: 6 args.
// ---------------------------------------------------------------------------
__device__ __forceinline__ void tdm_load_2d_bf16(uint32_t lds_addr,
                                                 const void* gtile,
                                                 uint32_t tile_d0, uint32_t tile_d1,
                                                 uint32_t stride_elems,
                                                 uint32_t tensor_d0, uint32_t tensor_d1) {
    uint64_t ga = (uint64_t)(uintptr_t)gtile;
    u32x4 g0;
    g0.x = 1u;                                                  // count=1, user desc
    g0.y = lds_addr;                                            // lds_addr[31:0]
    g0.z = (uint32_t)ga;                                        // global_addr[31:0]
    g0.w = (uint32_t)((ga >> 32) & 0x1FFFFFFu) | (2u << 30);    // addr[56:32] | type=2
    i32x8 g1;
    g1[0] = (int)((1u << 16)      // data_size = 2 bytes
                | (1u << 20)      // pad_enable
                | (3u << 22)      // pad_interval: 16 DWORDs
                | (3u << 25));    // pad_amount: 4 DWORDs
    g1[1] = (int)((tensor_d0 & 0xFFFFu) << 16);                         // dim0[15:0]
    g1[2] = (int)(((tensor_d0 >> 16) & 0xFFFFu) | ((tensor_d1 & 0xFFFFu) << 16));
    g1[3] = (int)(((tensor_d1 >> 16) & 0xFFFFu) | ((tile_d0 & 0xFFFFu) << 16));
    g1[4] = (int)(tile_d1 & 0xFFFFu);                                   // tile_dim1
    g1[5] = (int)stride_elems;                                          // dim0_stride[31:0]
    g1[6] = 0;                                                          // stride hi / dim1_stride lo
    g1[7] = 0;
    i32x4 z4 = {0, 0, 0, 0};
    i32x8 z8 = {0, 0, 0, 0, 0, 0, 0, 0};
    __builtin_amdgcn_tensor_load_to_lds(g0, g1, z4, z4, z8, 0);
}
__device__ __forceinline__ void wait_tensor0() {
    __builtin_amdgcn_s_wait_tensorcnt(0);
}

// ---------------------------------------------------------------------------
// fp32 -> bf16 conversion (grid-stride)
// ---------------------------------------------------------------------------
__global__ void cvt_f32_bf16_kernel(const float* __restrict__ in,
                                    __hip_bfloat16* __restrict__ out, size_t n) {
    size_t i = (size_t)blockIdx.x * blockDim.x + threadIdx.x;
    size_t stride = (size_t)gridDim.x * blockDim.x;
    for (; i < n; i += stride) out[i] = __float2bfloat16(in[i]);
}

// ---------------------------------------------------------------------------
// Generic bf16 WMMA GEMM:  C[M,N] = A[M,K] * W[N,K]^T + bias[N]
// Block tile 128x128, K-step 32, double-buffered TDM staging (one
// tensor_load_to_lds per tile, padded in-flight to LDS row stride 40).
// 8 waves (4x2), each computes 32x64 = 2x4 WMMA tiles -> 8 WMMA per K-step.
// ---------------------------------------------------------------------------
template<int EPI>
__global__ __launch_bounds__(256)
void gemm_bf16_kernel(const __hip_bfloat16* __restrict__ A,
                      const __hip_bfloat16* __restrict__ W,
                      const float* __restrict__ bias,
                      float* __restrict__ outF,
                      __hip_bfloat16* __restrict__ outB,
                      int M, int N, int K)
{
    constexpr int BM = 128, BN = 128, BK = 32, LDT = 40; // padded row stride
    __shared__ __hip_bfloat16 sA[2][BM * LDT];
    __shared__ __hip_bfloat16 sB[2][BN * LDT];

    const int t      = threadIdx.x;
    const int wave   = t >> 5;
    const int lane   = t & 31;
    const int lane16 = lane & 15;
    const int hi     = lane >> 4;
    const int blockM = blockIdx.y * BM;
    const int blockN = blockIdx.x * BN;
    const int WR     = (wave >> 1) * 32;   // wave row base (4 rows of waves)
    const int WC     = (wave & 1) * 64;    // wave col base (2 cols of waves)

    f32x8 acc[2][4];
    #pragma unroll
    for (int i = 0; i < 2; ++i)
        #pragma unroll
        for (int j = 0; j < 4; ++j)
            #pragma unroll
            for (int e = 0; e < 8; ++e) acc[i][j][e] = 0.0f;

    // one wave issues the A-tile DMA, another the B-tile DMA (EXEC ignored,
    // per-wave issue; TENSORcnt-tracked)
    auto stage = [&](int buf, int k0) {
        if (wave == 0) {
            tdm_load_2d_bf16(lds_offset_of(&sA[buf][0]),
                             A + (size_t)blockM * K + k0,
                             BK, BM, (uint32_t)K, (uint32_t)K, (uint32_t)M);
        } else if (wave == 1) {
            tdm_load_2d_bf16(lds_offset_of(&sB[buf][0]),
                             W + (size_t)blockN * K + k0,
                             BK, BN, (uint32_t)K, (uint32_t)K, (uint32_t)N);
        }
    };

    stage(0, 0);
    wait_tensor0();
    __syncthreads();

    int buf = 0;
    for (int k0 = 0; k0 < K; k0 += BK) {
        if (k0 + BK < K) stage(buf ^ 1, k0 + BK);

        // ---- fragments: contiguous k = 16*hi + s, same order for A and B ----
        FragB16 af[2], bfm[4];
        #pragma unroll
        for (int mi = 0; mi < 2; ++mi) {
            int r  = WR + mi * 16 + lane16;
            int kb = hi ? 16 : 0;
            af[mi].u[0] = *(const uint4*)&sA[buf][r * LDT + kb];
            af[mi].u[1] = *(const uint4*)&sA[buf][r * LDT + kb + 8];
        }
        #pragma unroll
        for (int ni = 0; ni < 4; ++ni) {
            int n  = WC + ni * 16 + lane16;
            int kb = hi ? 16 : 0;
            bfm[ni].u[0] = *(const uint4*)&sB[buf][n * LDT + kb];
            bfm[ni].u[1] = *(const uint4*)&sB[buf][n * LDT + kb + 8];
        }
        #pragma unroll
        for (int mi = 0; mi < 2; ++mi)
            #pragma unroll
            for (int ni = 0; ni < 4; ++ni)
                acc[mi][ni] = __builtin_amdgcn_wmma_f32_16x16x32_bf16(
                    false, af[mi].v, false, bfm[ni].v, (short)0, acc[mi][ni], false, false);

        wait_tensor0();      // issuing waves: next-buffer DMA complete
        __syncthreads();     // whole WG: data visible / safe to reuse old buf
        buf ^= 1;
    }

    // ---- epilogue: C layout = lane16 -> col, VGPR j -> row j + 8*hi ----
    #pragma unroll
    for (int mi = 0; mi < 2; ++mi) {
        #pragma unroll
        for (int ni = 0; ni < 4; ++ni) {
            int gn = blockN + WC + ni * 16 + lane16;
            float bv = bias[gn];
            #pragma unroll
            for (int jv = 0; jv < 8; ++jv) {
                int gm = blockM + WR + mi * 16 + jv + hi * 8;
                float val = acc[mi][ni][jv] + bv;
                if (EPI == EPI_F32) {
                    outF[(size_t)gm * N + gn] = val;
                } else if (EPI == EPI_GELU) {
                    outB[(size_t)gm * N + gn] = __float2bfloat16(gelu_exact(val));
                } else { // EPI_QKV: scatter into q/k/v [B*H, S, HD] bf16
                    int part = gn >> 10, d = gn & 1023;
                    int h = d >> 6, hd = d & 63;
                    int b = gm >> 6, s2 = gm & 63;
                    size_t idx = (size_t)part * QSZ +
                                 (((size_t)(b * NH + h) * SEQ + s2) * HD + hd);
                    outB[idx] = __float2bfloat16(val);
                }
            }
        }
    }
}

// ---------------------------------------------------------------------------
// Fused attention for one (b,h): scores = (q k^T)/8 + rel_bias, softmax, P*v
// q,k,v in [B*H, S, HD] bf16; ctx written as bf16 in [B, S, D] order.
// q,k staged via async LDS copies; v transposed through VGPRs.
// ---------------------------------------------------------------------------
__global__ __launch_bounds__(256)
void attention_kernel(const __hip_bfloat16* __restrict__ q,
                      const __hip_bfloat16* __restrict__ k,
                      const __hip_bfloat16* __restrict__ v,
                      const float* __restrict__ rel_bias,
                      __hip_bfloat16* __restrict__ ctx)
{
    constexpr int LD = 72, LDS_F = 68;
    __shared__ __hip_bfloat16 sQ[SEQ * LD];
    __shared__ __hip_bfloat16 sK[SEQ * LD];
    __shared__ __hip_bfloat16 sVT[HD * LD];   // v transposed: [hd][s]
    __shared__ __hip_bfloat16 sP[SEQ * LD];   // softmax probs (bf16)
    __shared__ float sS[SEQ * LDS_F];         // raw scores / exp values

    const int t  = threadIdx.x;
    const int bh = blockIdx.x;
    const int b  = bh >> 4, h = bh & 15;
    const size_t base = (size_t)bh * SEQ * HD;

    // ---- stage q, k (async to LDS); v transposed via VGPRs ----
    {
        int row = t >> 2, c0 = (t & 3) * 16;
        uint32_t goff = (uint32_t)((base + row * HD + c0) * 2);
        uint32_t lQ = lds_offset_of(&sQ[row * LD + c0]);
        uint32_t lK = lds_offset_of(&sK[row * LD + c0]);
        async_copy_b128(lQ,      q, goff);
        async_copy_b128(lQ + 16, q, goff + 16);
        async_copy_b128(lK,      k, goff);
        async_copy_b128(lK + 16, k, goff + 16);
        union { uint4 u; __hip_bfloat16 e[8]; } v0, v1;
        v0.u = *(const uint4*)(v + base + row * HD + c0);
        v1.u = *(const uint4*)(v + base + row * HD + c0 + 8);
        #pragma unroll
        for (int i = 0; i < 8; ++i) sVT[(c0 + i) * LD + row] = v0.e[i];
        #pragma unroll
        for (int i = 0; i < 8; ++i) sVT[(c0 + 8 + i) * LD + row] = v1.e[i];
    }
    wait_async0();
    __syncthreads();

    const int wave = t >> 5, lane = t & 31, lane16 = lane & 15, hi = lane >> 4;

    // ---- scores: 4x4 grid of 16x16 tiles, each wave handles 2 ----
    for (int ti = wave * 2; ti < wave * 2 + 2; ++ti) {
        int mi = ti >> 2, ni = ti & 3;
        f32x8 acc;
        #pragma unroll
        for (int e = 0; e < 8; ++e) acc[e] = 0.0f;
        #pragma unroll
        for (int ks = 0; ks < HD; ks += 32) {
            FragB16 af, bfm;
            int r  = mi * 16 + lane16, kb = ks + (hi ? 16 : 0);
            af.u[0] = *(const uint4*)&sQ[r * LD + kb];
            af.u[1] = *(const uint4*)&sQ[r * LD + kb + 8];
            int n = ni * 16 + lane16, kb2 = ks + (hi ? 16 : 0);
            bfm.u[0] = *(const uint4*)&sK[n * LD + kb2];
            bfm.u[1] = *(const uint4*)&sK[n * LD + kb2 + 8];
            acc = __builtin_amdgcn_wmma_f32_16x16x32_bf16(
                false, af.v, false, bfm.v, (short)0, acc, false, false);
        }
        int gn = ni * 16 + lane16;
        #pragma unroll
        for (int jv = 0; jv < 8; ++jv) {
            int gm = mi * 16 + jv + hi * 8;
            sS[gm * LDS_F + gn] =
                acc[jv] * 0.125f + rel_bias[((size_t)h * SEQ + gm) * SEQ + gn];
        }
    }
    __syncthreads();

    // ---- softmax per row (one thread per row) ----
    if (t < SEQ) {
        float mx = -3.4e38f;
        for (int j = 0; j < SEQ; ++j) mx = fmaxf(mx, sS[t * LDS_F + j]);
        float sum = 0.0f;
        for (int j = 0; j < SEQ; ++j) {
            float e = __expf(sS[t * LDS_F + j] - mx);
            sS[t * LDS_F + j] = e;
            sum += e;
        }
        float inv = 1.0f / sum;
        for (int j = 0; j < SEQ; ++j)
            sP[t * LD + j] = __float2bfloat16(sS[t * LDS_F + j] * inv);
    }
    __syncthreads();

    // ---- ctx = P @ V : B-matrix rows come from sVT (contiguous over s) ----
    for (int ti = wave * 2; ti < wave * 2 + 2; ++ti) {
        int mi = ti >> 2, ni = ti & 3;
        f32x8 acc;
        #pragma unroll
        for (int e = 0; e < 8; ++e) acc[e] = 0.0f;
        #pragma unroll
        for (int ks = 0; ks < SEQ; ks += 32) {
            FragB16 af, bfm;
            int r  = mi * 16 + lane16, kb = ks + (hi ? 16 : 0);
            af.u[0] = *(const uint4*)&sP[r * LD + kb];
            af.u[1] = *(const uint4*)&sP[r * LD + kb + 8];
            int n = ni * 16 + lane16, kb2 = ks + (hi ? 16 : 0);
            bfm.u[0] = *(const uint4*)&sVT[n * LD + kb2];
            bfm.u[1] = *(const uint4*)&sVT[n * LD + kb2 + 8];
            acc = __builtin_amdgcn_wmma_f32_16x16x32_bf16(
                false, af.v, false, bfm.v, (short)0, acc, false, false);
        }
        int hd = ni * 16 + lane16;
        #pragma unroll
        for (int jv = 0; jv < 8; ++jv) {
            int i = mi * 16 + jv + hi * 8;
            ctx[((size_t)(b * SEQ + i) * DIM) + h * HD + hd] = __float2bfloat16(acc[jv]);
        }
    }
}

// ---------------------------------------------------------------------------
// y = LayerNorm(x + t) * gamma + beta; writes fp32 (+ optional bf16 copy)
// One block per 1024-wide row.
// ---------------------------------------------------------------------------
__global__ __launch_bounds__(256)
void add_ln_kernel(const float* __restrict__ x, const float* __restrict__ tt,
                   const float* __restrict__ gamma, const float* __restrict__ beta,
                   float* __restrict__ yF, __hip_bfloat16* __restrict__ yB)
{
    __shared__ float r1[256], r2[256];
    const int row = blockIdx.x, t = threadIdx.x;
    const size_t off = (size_t)row * DIM;
    float vals[4];
    float s = 0.0f, sq = 0.0f;
    #pragma unroll
    for (int i = 0; i < 4; ++i) {
        int c = t + i * 256;
        float vv = x[off + c] + tt[off + c];
        vals[i] = vv; s += vv; sq += vv * vv;
    }
    r1[t] = s; r2[t] = sq;
    __syncthreads();
    for (int o = 128; o > 0; o >>= 1) {
        if (t < o) { r1[t] += r1[t + o]; r2[t] += r2[t + o]; }
        __syncthreads();
    }
    float mean = r1[0] * (1.0f / DIM);
    float var  = r2[0] * (1.0f / DIM) - mean * mean;
    float rstd = rsqrtf(var + 1e-5f);
    #pragma unroll
    for (int i = 0; i < 4; ++i) {
        int c = t + i * 256;
        float y = (vals[i] - mean) * rstd * gamma[c] + beta[c];
        yF[off + c] = y;
        if (yB) yB[off + c] = __float2bfloat16(y);
    }
}

// ---------------------------------------------------------------------------
// Host-side orchestration
// ---------------------------------------------------------------------------
extern "C" void kernel_launch(void* const* d_in, const int* in_sizes, int n_in,
                              void* d_out, int out_size, void* d_ws, size_t ws_size,
                              hipStream_t stream) {
    const float* x     = (const float*)d_in[0];
    const float* w_in  = (const float*)d_in[1];
    const float* b_in  = (const float*)d_in[2];
    const float* w_out = (const float*)d_in[3];
    const float* b_out = (const float*)d_in[4];
    const float* rel   = (const float*)d_in[5];
    const float* g1    = (const float*)d_in[6];
    const float* be1   = (const float*)d_in[7];
    const float* w1    = (const float*)d_in[8];
    const float* b1    = (const float*)d_in[9];
    const float* w2    = (const float*)d_in[10];
    const float* b2    = (const float*)d_in[11];
    const float* g2    = (const float*)d_in[12];
    const float* be2   = (const float*)d_in[13];

    // Workspace layout (256B aligned regions)
    char* base = (char*)d_ws;
    size_t off = 0;
    auto alloc = [&](size_t bytes) -> char* {
        char* p = base + off;
        off = (off + bytes + 255) & ~(size_t)255;
        return p;
    };
    __hip_bfloat16* xb     = (__hip_bfloat16*)alloc((size_t)MTOK * DIM * 2);
    __hip_bfloat16* w_in_b = (__hip_bfloat16*)alloc((size_t)3 * DIM * DIM * 2);
    __hip_bfloat16* w_outb = (__hip_bfloat16*)alloc((size_t)DIM * DIM * 2);
    __hip_bfloat16* w1b    = (__hip_bfloat16*)alloc((size_t)FFN * DIM * 2);
    __hip_bfloat16* w2b    = (__hip_bfloat16*)alloc((size_t)DIM * FFN * 2);
    __hip_bfloat16* qkvb   = (__hip_bfloat16*)alloc(3 * QSZ * 2);  // q | k | v
    __hip_bfloat16* ctxb   = (__hip_bfloat16*)alloc((size_t)MTOK * DIM * 2);
    float*          attn_o = (float*)alloc((size_t)MTOK * DIM * 4);
    float*          x1f    = (float*)alloc((size_t)MTOK * DIM * 4);
    __hip_bfloat16* x1b    = (__hip_bfloat16*)alloc((size_t)MTOK * DIM * 2);
    __hip_bfloat16* hb     = (__hip_bfloat16*)alloc((size_t)MTOK * FFN * 2);
    float*          ffn_o  = (float*)alloc((size_t)MTOK * DIM * 4);

    // 1) Convert x + weights to bf16
    cvt_f32_bf16_kernel<<<2048, 256, 0, stream>>>(x,     xb,     (size_t)MTOK * DIM);
    cvt_f32_bf16_kernel<<<2048, 256, 0, stream>>>(w_in,  w_in_b, (size_t)3 * DIM * DIM);
    cvt_f32_bf16_kernel<<<1024, 256, 0, stream>>>(w_out, w_outb, (size_t)DIM * DIM);
    cvt_f32_bf16_kernel<<<2048, 256, 0, stream>>>(w1,    w1b,    (size_t)FFN * DIM);
    cvt_f32_bf16_kernel<<<2048, 256, 0, stream>>>(w2,    w2b,    (size_t)DIM * FFN);

    // 2) QKV projection (scatter epilogue into per-head q/k/v)
    gemm_bf16_kernel<EPI_QKV><<<dim3(3 * DIM / 128, MTOK / 128), 256, 0, stream>>>(
        xb, w_in_b, b_in, nullptr, qkvb, MTOK, 3 * DIM, DIM);

    // 3) Attention per (b,h)
    attention_kernel<<<BSZ * NH, 256, 0, stream>>>(
        qkvb, qkvb + QSZ, qkvb + 2 * QSZ, rel, ctxb);

    // 4) Output projection
    gemm_bf16_kernel<EPI_F32><<<dim3(DIM / 128, MTOK / 128), 256, 0, stream>>>(
        ctxb, w_outb, b_out, attn_o, nullptr, MTOK, DIM, DIM);

    // 5) x1 = LN(x + attn_out)
    add_ln_kernel<<<MTOK, 256, 0, stream>>>(x, attn_o, g1, be1, x1f, x1b);

    // 6) h = gelu(x1 @ w1^T + b1), stored bf16
    gemm_bf16_kernel<EPI_GELU><<<dim3(FFN / 128, MTOK / 128), 256, 0, stream>>>(
        x1b, w1b, b1, nullptr, hb, MTOK, FFN, DIM);

    // 7) ffn = h @ w2^T + b2
    gemm_bf16_kernel<EPI_F32><<<dim3(DIM / 128, MTOK / 128), 256, 0, stream>>>(
        hb, w2b, b2, ffn_o, nullptr, MTOK, DIM, FFN);

    // 8) out = LN(x1 + ffn)
    add_ln_kernel<<<MTOK, 256, 0, stream>>>(x1f, ffn_o, g2, be2, (float*)d_out, nullptr);
}
